// _SingleDirectionISSMLayer_17222818857095
// MI455X (gfx1250) — compile-verified, matrix operand order checked by reading, not run on hardware
//
#include <hip/hip_runtime.h>
#include <stdint.h>

#define BSZ   2
#define NQ    128
#define LSEQ  2048
#define DM    256
#define DI    512
#define HH    8
#define HD    64
#define DDIST 16
#define DCONV 4
#define DHID  64
#define DKEY  1034
#define DKEYP 1040   // padded to multiple of 16 for WMMA N tiles
#define DCV   514
#define CH    16     // scan chunk (timesteps per TDM transfer)

typedef __bf16 bf16_t;
typedef bf16_t   v16bf __attribute__((ext_vector_type(16)));
typedef float    v8f   __attribute__((ext_vector_type(8)));
typedef uint32_t u32x4 __attribute__((ext_vector_type(4)));
typedef uint32_t u32x8 __attribute__((ext_vector_type(8)));

union FragB { v16bf v; uint32_t u[8]; };

__device__ __forceinline__ uint16_t f2bf(float f) {
  uint32_t u = __float_as_uint(f);
  uint32_t r = (u + 0x7FFFu + ((u >> 16) & 1u)) >> 16;
  return (uint16_t)r;
}
__device__ __forceinline__ float sigmoidf_(float x) { return 1.0f / (1.0f + __expf(-x)); }

// ---------------------------------------------------------------------------
// Tensor Data Mover: build D# (ISA 08_async_tensor §8) and issue a DMA into LDS.
// Two descriptor groups -> tensors up to 2D.  data_size = 4 bytes (code 2).
__device__ __forceinline__ void tdm_load_1d(const float* g, uint32_t lds_byte, uint32_t n) {
  uint64_t ga = (uint64_t)(uintptr_t)g;
  u32x4 g0;
  g0[0] = 1u;                                             // count=1, user mode
  g0[1] = lds_byte;                                       // lds_addr
  g0[2] = (uint32_t)ga;                                   // global_addr[31:0]
  g0[3] = (uint32_t)((ga >> 32) & 0x01FFFFFFu) | (2u << 30); // addr[56:32] | type=2
  u32x8 g1;
  g1[0] = 2u << 16;                                       // data_size = 4B
  g1[1] = (n & 0xFFFFu) << 16;                            // tensor_dim0 [15:0]
  g1[2] = (n >> 16) & 0xFFFFu;                            // tensor_dim0 [31:16], dim1=0
  g1[3] = (n & 0xFFFFu) << 16;                            // tile_dim0 = n
  g1[4] = 0u;                                             // tile_dim1/2 = 0
  g1[5] = 0u; g1[6] = 0u; g1[7] = 0u;                     // strides unused for 1D
  asm volatile("tensor_load_to_lds %0, %1" :: "s"(g0), "s"(g1) : "memory");
}

__device__ __forceinline__ void tdm_load_2d(const float* g, uint32_t lds_byte,
                                            uint32_t tile_w, uint32_t rows, uint32_t stride) {
  uint64_t ga = (uint64_t)(uintptr_t)g;
  u32x4 g0;
  g0[0] = 1u;
  g0[1] = lds_byte;
  g0[2] = (uint32_t)ga;
  g0[3] = (uint32_t)((ga >> 32) & 0x01FFFFFFu) | (2u << 30);
  u32x8 g1;
  g1[0] = 2u << 16;                                       // data_size = 4B
  g1[1] = (stride & 0xFFFFu) << 16;                       // tensor_dim0 = row length
  g1[2] = ((stride >> 16) & 0xFFFFu) | ((rows & 0xFFFFu) << 16); // dim0 hi | tensor_dim1 lo
  g1[3] = ((rows >> 16) & 0xFFFFu) | (tile_w << 16);      // tensor_dim1 hi | tile_dim0
  g1[4] = rows & 0xFFFFu;                                 // tile_dim1 = rows
  g1[5] = stride;                                         // tensor_dim0_stride lo32
  g1[6] = 0u; g1[7] = 0u;
  asm volatile("tensor_load_to_lds %0, %1" :: "s"(g0), "s"(g1) : "memory");
}

// ---------------------------------------------------------------------------
// elementwise add + convert to bf16 (for WMMA A operands)
__global__ void k_add_bf16(const float* __restrict__ a, const float* __restrict__ b,
                           uint16_t* __restrict__ out, int n) {
  int i = blockIdx.x * blockDim.x + threadIdx.x;
  if (i < n) out[i] = f2bf(a[i] + b[i]);
}

// transpose + convert weight W (K x N) -> Wt (Npad x K) bf16, zero-padded cols
__global__ void k_wconv(const float* __restrict__ W, uint16_t* __restrict__ Wt,
                        int K, int N, int Npad) {
  int i = blockIdx.x * blockDim.x + threadIdx.x;
  if (i >= K * Npad) return;
  int k = i % K, n = i / K;
  Wt[(size_t)n * K + k] = (n < N) ? f2bf(W[(size_t)k * N + n]) : (uint16_t)0;
}

// ---------------------------------------------------------------------------
// WMMA bf16 GEMM: C(MxN) = A(MxK) * Bt(NxK)^T  (+bias[n]) (+residual[m,n])
__global__ void __launch_bounds__(256)
k_gemm_bf16(const uint16_t* __restrict__ A, const uint16_t* __restrict__ Bt,
            float* __restrict__ C, int M, int N, int K,
            const float* __restrict__ bias, const float* __restrict__ residual,
            int total_tiles) {
  int wid = (blockIdx.x * blockDim.x + threadIdx.x) >> 5;
  if (wid >= total_tiles) return;
  int lane = threadIdx.x & 31;
  int ntiles = N >> 4;
  int mt = wid / ntiles, nt = wid % ntiles;
  int lr = lane & 15, lh = lane >> 4;
  const uint16_t* Arow = A + (size_t)(mt * 16 + lr) * K;
  const uint16_t* Brow = Bt + (size_t)(nt * 16 + lr) * K;
  v8f acc = {0.f, 0.f, 0.f, 0.f, 0.f, 0.f, 0.f, 0.f};
  for (int k0 = 0; k0 < K; k0 += 32) {
    FragB a, b;
#pragma unroll
    for (int v = 0; v < 8; ++v) {
      // 16-bit A layout: lanes 0-15 rows, VGPR v holds K = (v/4)*16 + (lane/16)*8 + (v%4)*2
      int k = k0 + ((v >> 2) << 4) + (lh << 3) + ((v & 3) << 1);
      a.u[v] = *(const uint32_t*)(Arow + k);
      b.u[v] = *(const uint32_t*)(Brow + k);
    }
    acc = __builtin_amdgcn_wmma_f32_16x16x32_bf16(false, a.v, false, b.v,
                                                  (short)0, acc, false, false);
  }
  int n = nt * 16 + lr;
  float bv = bias ? bias[n] : 0.0f;
#pragma unroll
  for (int r = 0; r < 8; ++r) {
    int m = mt * 16 + lh * 8 + r;   // C/D layout: VGPR r, lane-half selects M=r / M=8+r
    size_t idx = (size_t)m * N + n;
    float out = acc[r] + bv;
    if (residual) out += residual[idx];
    C[idx] = out;
  }
}

// ---------------------------------------------------------------------------
// causal depthwise conv over xbc channels of kp + SiLU -> x, b_bias, c_bias
__global__ void k_dwconv(const float* __restrict__ kp, const float* __restrict__ wconv,
                         const float* __restrict__ bconv, float* __restrict__ x,
                         float* __restrict__ bbias, float* __restrict__ cbias) {
  int i = blockIdx.x * blockDim.x + threadIdx.x;
  if (i >= BSZ * LSEQ * DCV) return;
  int c = i % DCV;
  int l = (i / DCV) % LSEQ;
  int b = i / (DCV * LSEQ);
  const float* col = kp + (size_t)(b * LSEQ) * DKEYP + (DI + c);
  float acc = bconv[c];
#pragma unroll
  for (int t = 0; t < DCONV; ++t) {
    int ls = l - (DCONV - 1) + t;
    if (ls >= 0) acc += col[(size_t)ls * DKEYP] * wconv[c * DCONV + t];
  }
  float s = acc * sigmoidf_(acc);
  size_t bl = (size_t)(b * LSEQ + l);
  if (c < DI)      x[bl * DI + c] = s;
  else if (c == DI) bbias[bl] = s;
  else              cbias[bl] = s;
}

// ---------------------------------------------------------------------------
// distance MLP -> Cssm, decay = exp(dt*A), dtB = dt*B  (all transcendentals here)
__global__ void __launch_bounds__(256)
k_dist(const float* __restrict__ anchors, const float* __restrict__ keypos,
       const float* __restrict__ kp, const float* __restrict__ bbias,
       const float* __restrict__ cbias,
       const float* __restrict__ w1, const float* __restrict__ b1,
       const float* __restrict__ w2, const float* __restrict__ bcw,
       const float* __restrict__ dtw, const float* __restrict__ dtbias,
       const float* __restrict__ alog,
       float* __restrict__ Cssm, float* __restrict__ decay, float* __restrict__ dtB) {
  __shared__ float s_w1[3 * DHID], s_b1[DHID], s_w2[DHID * DDIST], s_bc[DDIST * 2],
                   s_dtw[DDIST * HH], s_dtb[HH], s_negA[HH];
  int tid = threadIdx.x;
  if (tid < 3 * DHID) s_w1[tid] = w1[tid];
  if (tid < DHID) s_b1[tid] = b1[tid];
  for (int j = tid; j < DHID * DDIST; j += 256) s_w2[j] = w2[j];
  if (tid < DDIST * 2) s_bc[tid] = bcw[tid];
  if (tid < DDIST * HH) s_dtw[tid] = dtw[tid];
  if (tid < HH) { s_dtb[tid] = dtbias[tid]; s_negA[tid] = -__expf(alog[tid]); }
  __syncthreads();

  int i = blockIdx.x * 256 + tid;
  if (i >= BSZ * LSEQ * NQ) return;
  int n = i % NQ;
  int l = (i / NQ) % LSEQ;
  int b = i / (NQ * LSEQ);

  float enc[3];
#pragma unroll
  for (int d = 0; d < 3; ++d) {
    float delta = anchors[(size_t)(b * NQ + n) * 3 + d] - keypos[(size_t)(b * LSEQ + l) * 3 + d];
    float e = __log2f(fabsf(delta) * 20.0f + 1.0f) * (1.0f / 12.0f);  // /log2(8)/4
    enc[d] = (delta < 0.0f) ? -e : e;
  }
  float feat[DDIST];
#pragma unroll
  for (int q = 0; q < DDIST; ++q) feat[q] = 0.0f;
  for (int j = 0; j < DHID; ++j) {
    float hval = s_b1[j] + enc[0] * s_w1[j] + enc[1] * s_w1[DHID + j] + enc[2] * s_w1[2 * DHID + j];
    hval = fmaxf(hval, 0.0f);
#pragma unroll
    for (int q = 0; q < DDIST; ++q) feat[q] += hval * s_w2[j * DDIST + q];
  }
  float bc0 = 0.0f, bc1 = 0.0f;
#pragma unroll
  for (int q = 0; q < DDIST; ++q) { bc0 += feat[q] * s_bc[q * 2]; bc1 += feat[q] * s_bc[q * 2 + 1]; }
  size_t bl = (size_t)(b * LSEQ + l);
  float Bv = bc0 + bbias[bl];
  Cssm[bl * NQ + n] = bc1 + cbias[bl];
  const float* dtseq = kp + bl * DKEYP + (DI + DCV);   // dtb_seq columns 1026..1033
#pragma unroll
  for (int h = 0; h < HH; ++h) {
    float dv = 0.0f;
#pragma unroll
    for (int q = 0; q < DDIST; ++q) dv += feat[q] * s_dtw[q * HH + h];
    float z = dv + dtseq[h] + s_dtb[h];
    float dtf = (z > 20.0f) ? z : log1pf(__expf(z));
    size_t idx = ((size_t)((b * HH + h) * LSEQ + l)) * NQ + n;
    decay[idx] = __expf(dtf * s_negA[h]);
    dtB[idx] = dtf * Bv;
  }
}

// ---------------------------------------------------------------------------
// serial selective scan; one workgroup per (dir, b, h); state in VGPRs.
// Per-chunk coefficients (decay/dtB/C 1-D tiles, x 2-D tile) are DMAed into a
// double-buffered LDS region by the Tensor Data Mover (wave 0 issues, tracked
// with TENSORcnt), overlapping the DMA of chunk c+1 with compute of chunk c.
__global__ void __launch_bounds__(256)
k_scan(const float* __restrict__ decay, const float* __restrict__ dtB,
       const float* __restrict__ Cssm, const float* __restrict__ x,
       const float* __restrict__ init,
       float* __restrict__ ysf, float* __restrict__ ysb, float* __restrict__ hstate) {
  // buffer: decay[CH*NQ] | dtB[CH*NQ] | C[CH*NQ] | x[CH*HD]
  __shared__ float sb[2][3 * CH * NQ + CH * HD];
  __shared__ float s_y[HD * 4];
  int wg = blockIdx.x;               // dir*16 + b*8 + h
  int dir = wg >> 4;
  int b = (wg >> 3) & 1;
  int h = wg & 7;
  int tid = threadIdx.x;
  int sub = tid >> 6;                // quarter of the n-dim
  int p = tid & 63;
  int nb = sub * 32;

  float S[32];
#pragma unroll
  for (int i = 0; i < 32; ++i)
    S[i] = init[(size_t)(b * NQ + nb + i) * DI + h * HD + p];

  const float* dec = decay + (size_t)((b * HH + h) * LSEQ) * NQ;
  const float* dtb = dtB + (size_t)((b * HH + h) * LSEQ) * NQ;
  const float* Cp = Cssm + (size_t)(b * LSEQ) * NQ;
  const float* xp = x + (size_t)(b * LSEQ) * DI + h * HD;
  float* ys = (dir ? ysb : ysf) + (size_t)(b * LSEQ) * DI + h * HD;

  const int NCH = LSEQ / CH;
  auto issue = [&](int c) {
    int tlo = dir ? (LSEQ - CH * (c + 1)) : (c * CH);
    uint32_t base = (uint32_t)(uintptr_t)(&sb[c & 1][0]);
    tdm_load_1d(dec + (size_t)tlo * NQ, base, CH * NQ);
    tdm_load_1d(dtb + (size_t)tlo * NQ, base + CH * NQ * 4u, CH * NQ);
    tdm_load_1d(Cp + (size_t)tlo * NQ, base + 2u * CH * NQ * 4u, CH * NQ);
    tdm_load_2d(xp + (size_t)tlo * DI, base + 3u * CH * NQ * 4u, HD, CH, DI);
  };

  if (tid < 32) issue(0);
  for (int c = 0; c < NCH; ++c) {
    if (tid < 32) {
      if (c + 1 < NCH) {
        issue(c + 1);
        __builtin_amdgcn_s_wait_tensorcnt(4);   // chunk c complete (in-order)
      } else {
        __builtin_amdgcn_s_wait_tensorcnt(0);
      }
    }
    __syncthreads();
    const float* B = sb[c & 1];
    int tlo = dir ? (LSEQ - CH * (c + 1)) : (c * CH);
    for (int rr = 0; rr < CH; ++rr) {
      int r = dir ? (CH - 1 - rr) : rr;
      float xv = B[3 * CH * NQ + r * HD + p];
      float acc = 0.0f;
#pragma unroll
      for (int i = 0; i < 32; ++i) {
        S[i] = S[i] * B[r * NQ + nb + i] + B[CH * NQ + r * NQ + nb + i] * xv;
        acc += S[i] * B[2 * CH * NQ + r * NQ + nb + i];
      }
      s_y[p * 4 + sub] = acc;
      __syncthreads();
      if (sub == 0)
        ys[(size_t)(tlo + r) * DI + p] =
            s_y[p * 4] + s_y[p * 4 + 1] + s_y[p * 4 + 2] + s_y[p * 4 + 3];
      __syncthreads();
    }
  }
#pragma unroll
  for (int i = 0; i < 32; ++i)
    hstate[(size_t)(((dir * BSZ + b) * HH + h) * HD + p) * NQ + nb + i] = S[i];
}

// ---------------------------------------------------------------------------
// y combine + D skip + SiLU gate + RMSNorm -> bf16 (A of out_key_proj GEMM)
__global__ void __launch_bounds__(256)
k_gate(const float* __restrict__ ysf, const float* __restrict__ ysb,
       const float* __restrict__ x, const float* __restrict__ kp,
       const float* __restrict__ Dp, const float* __restrict__ knw,
       uint16_t* __restrict__ gout) {
  __shared__ float red[256];
  int bl = blockIdx.x;
  int tid = threadIdx.x;
  size_t base = (size_t)bl * DI;
  const float* kprow = kp + (size_t)bl * DKEYP;
  float g[2];
  float ss = 0.0f;
#pragma unroll
  for (int e = 0; e < 2; ++e) {
    int di = tid + e * 256;
    float xv = x[base + di];
    // y = 0.5*(ys_f + ys_b) + D*x (inside issm, once after averaging) + D*x (outer)
    float yv = 0.5f * (ysf[base + di] + ysb[base + di]) + 2.0f * Dp[di >> 6] * xv;
    float z = kprow[di];
    float gv = yv * (z * sigmoidf_(z));
    g[e] = gv;
    ss += gv * gv;
  }
  red[tid] = ss;
  __syncthreads();
  for (int o = 128; o > 0; o >>= 1) { if (tid < o) red[tid] += red[tid + o]; __syncthreads(); }
  float rms = rsqrtf(red[0] * (1.0f / DI) + 1e-5f);
#pragma unroll
  for (int e = 0; e < 2; ++e) {
    int di = tid + e * 256;
    gout[base + di] = f2bf(g[e] * rms * knw[di]);
  }
}

// ---------------------------------------------------------------------------
// hs = 0.5*(h_f + h_b) gathered to (b,n,DI), LayerNorm -> bf16
__global__ void __launch_bounds__(256)
k_hs_ln(const float* __restrict__ hstate, const float* __restrict__ w,
        const float* __restrict__ bb, uint16_t* __restrict__ out) {
  __shared__ float red[256];
  int bn = blockIdx.x;
  int b = bn / NQ, n = bn % NQ;
  int tid = threadIdx.x;
  float v[2];
  float s1 = 0.0f;
#pragma unroll
  for (int e = 0; e < 2; ++e) {
    int di = tid + e * 256;
    int h = di >> 6, p = di & 63;
    size_t i0 = (size_t)((b * HH + h) * HD + p) * NQ + n;
    size_t i1 = i0 + (size_t)BSZ * HH * HD * NQ;
    v[e] = 0.5f * (hstate[i0] + hstate[i1]);
    s1 += v[e];
  }
  red[tid] = s1;
  __syncthreads();
  for (int o = 128; o > 0; o >>= 1) { if (tid < o) red[tid] += red[tid + o]; __syncthreads(); }
  float mean = red[0] * (1.0f / DI);
  __syncthreads();
  float s2 = 0.0f;
#pragma unroll
  for (int e = 0; e < 2; ++e) { float d = v[e] - mean; s2 += d * d; }
  red[tid] = s2;
  __syncthreads();
  for (int o = 128; o > 0; o >>= 1) { if (tid < o) red[tid] += red[tid + o]; __syncthreads(); }
  float inv = rsqrtf(red[0] * (1.0f / DI) + 1e-5f);
#pragma unroll
  for (int e = 0; e < 2; ++e) {
    int di = tid + e * 256;
    out[(size_t)bn * DI + di] = f2bf((v[e] - mean) * inv * w[di] + bb[di]);
  }
}

// LayerNorm over 256 cols -> bf16
__global__ void __launch_bounds__(256)
k_ln256(const float* __restrict__ X, const float* __restrict__ w,
        const float* __restrict__ bb, uint16_t* __restrict__ out) {
  __shared__ float red[256];
  int r = blockIdx.x;
  int tid = threadIdx.x;
  float v = X[(size_t)r * DM + tid];
  red[tid] = v;
  __syncthreads();
  for (int o = 128; o > 0; o >>= 1) { if (tid < o) red[tid] += red[tid + o]; __syncthreads(); }
  float mean = red[0] * (1.0f / DM);
  __syncthreads();
  float d = v - mean;
  red[tid] = d * d;
  __syncthreads();
  for (int o = 128; o > 0; o >>= 1) { if (tid < o) red[tid] += red[tid + o]; __syncthreads(); }
  float inv = rsqrtf(red[0] * (1.0f / DM) + 1e-5f);
  out[(size_t)r * DM + tid] = f2bf(d * inv * w[tid] + bb[tid]);
}

// act = gelu(a) * v -> bf16 (A of fc2 GEMM)
__global__ void k_gelumul(const float* __restrict__ t1, uint16_t* __restrict__ act) {
  int i = blockIdx.x * blockDim.x + threadIdx.x;
  if (i >= (BSZ * NQ) * (4 * DM)) return;
  int r = i / (4 * DM), j = i % (4 * DM);
  float a = t1[(size_t)r * (8 * DM) + j];
  float v = t1[(size_t)r * (8 * DM) + 4 * DM + j];
  float ge = 0.5f * a * (1.0f + erff(a * 0.70710678118654752f));
  act[i] = f2bf(ge * v);
}

// ---------------------------------------------------------------------------
extern "C" void kernel_launch(void* const* d_in, const int* in_sizes, int n_in,
                              void* d_out, int out_size, void* d_ws, size_t ws_size,
                              hipStream_t stream) {
  (void)in_sizes; (void)n_in; (void)out_size; (void)ws_size;
  const float* queries        = (const float*)d_in[0];
  const float* anchors        = (const float*)d_in[1];
  const float* features       = (const float*)d_in[2];
  const float* pos_embed      = (const float*)d_in[3];
  const float* key_pos        = (const float*)d_in[4];
  const float* query_pos      = (const float*)d_in[5];
  const float* key_proj_w     = (const float*)d_in[6];
  const float* key_conv_w     = (const float*)d_in[7];
  const float* key_conv_b     = (const float*)d_in[8];
  const float* query_proj_w   = (const float*)d_in[9];
  const float* dist_w1        = (const float*)d_in[10];
  const float* dist_b1        = (const float*)d_in[11];
  const float* dist_w2        = (const float*)d_in[12];
  const float* bc_proj_w      = (const float*)d_in[13];
  const float* dt_proj_w      = (const float*)d_in[14];
  const float* dt_bias        = (const float*)d_in[15];
  const float* A_log          = (const float*)d_in[16];
  const float* Dvec           = (const float*)d_in[17];
  const float* key_norm_w     = (const float*)d_in[18];
  const float* out_key_proj_w = (const float*)d_in[19];
  const float* query_norm_w   = (const float*)d_in[20];
  const float* query_norm_b   = (const float*)d_in[21];
  const float* out_query_proj_w = (const float*)d_in[22];
  const float* ffn_norm_w     = (const float*)d_in[23];
  const float* ffn_norm_b     = (const float*)d_in[24];
  const float* ffn_fc1_w      = (const float*)d_in[25];
  const float* ffn_fc1_b      = (const float*)d_in[26];
  const float* ffn_fc2_w      = (const float*)d_in[27];
  const float* ffn_fc2_b      = (const float*)d_in[28];

  float* out_q    = (float*)d_out;            // B*NQ*DM
  float* out_feat = (float*)d_out + BSZ * NQ * DM;

  uint8_t* wp = (uint8_t*)d_ws;
  auto alloc = [&](size_t bytes) -> void* {
    void* p = (void*)wp;
    wp += (bytes + 255) & ~(size_t)255;
    return p;
  };
  const size_t BL = (size_t)BSZ * LSEQ;
  uint16_t* fin_bf  = (uint16_t*)alloc(BL * DM * 2);
  uint16_t* qp_bf   = (uint16_t*)alloc((size_t)BSZ * NQ * DM * 2);
  uint16_t* Wkey_t  = (uint16_t*)alloc((size_t)DKEYP * DM * 2);
  uint16_t* Wq_t    = (uint16_t*)alloc((size_t)DI * DM * 2);
  uint16_t* Wok_t   = (uint16_t*)alloc((size_t)DM * DI * 2);
  uint16_t* Woq_t   = (uint16_t*)alloc((size_t)DM * DI * 2);
  uint16_t* W1_t    = (uint16_t*)alloc((size_t)(8 * DM) * DM * 2);
  uint16_t* W2_t    = (uint16_t*)alloc((size_t)DM * (4 * DM) * 2);
  float* kp     = (float*)alloc(BL * DKEYP * 4);
  float* xbuf   = (float*)alloc(BL * DI * 4);
  float* bbias  = (float*)alloc(BL * 4);
  float* cbias  = (float*)alloc(BL * 4);
  float* Cssm   = (float*)alloc(BL * NQ * 4);
  float* decay  = (float*)alloc((size_t)BSZ * HH * LSEQ * NQ * 4);
  float* dtB    = (float*)alloc((size_t)BSZ * HH * LSEQ * NQ * 4);
  float* initb  = (float*)alloc((size_t)BSZ * NQ * DI * 4);
  float* ysf    = (float*)alloc(BL * DI * 4);
  float* ysb    = (float*)alloc(BL * DI * 4);
  float* hstate = (float*)alloc((size_t)2 * BSZ * HH * HD * NQ * 4);
  uint16_t* g_bf    = (uint16_t*)alloc(BL * DI * 2);
  uint16_t* lnhs_bf = (uint16_t*)alloc((size_t)BSZ * NQ * DI * 2);
  float* outq_pre   = (float*)alloc((size_t)BSZ * NQ * DM * 4);
  uint16_t* fn_bf   = (uint16_t*)alloc((size_t)BSZ * NQ * DM * 2);
  float* t1         = (float*)alloc((size_t)BSZ * NQ * 8 * DM * 4);
  uint16_t* act_bf  = (uint16_t*)alloc((size_t)BSZ * NQ * 4 * DM * 2);

  auto gemm = [&](const uint16_t* A, const uint16_t* Bt, float* C, int M, int N, int K,
                  const float* bias, const float* residual) {
    int tiles = (M / 16) * (N / 16);
    int blocks = (tiles + 7) / 8;
    k_gemm_bf16<<<blocks, 256, 0, stream>>>(A, Bt, C, M, N, K, bias, residual, tiles);
  };

  // inputs -> bf16 operands
  { int n = (int)(BL * DM);
    k_add_bf16<<<(n + 255) / 256, 256, 0, stream>>>(features, pos_embed, fin_bf, n); }
  { int n = BSZ * NQ * DM;
    k_add_bf16<<<(n + 255) / 256, 256, 0, stream>>>(queries, query_pos, qp_bf, n); }
  // weight transposes
  k_wconv<<<(DM * DKEYP + 255) / 256, 256, 0, stream>>>(key_proj_w, Wkey_t, DM, DKEY, DKEYP);
  k_wconv<<<(DM * DI + 255) / 256, 256, 0, stream>>>(query_proj_w, Wq_t, DM, DI, DI);
  k_wconv<<<(DI * DM + 255) / 256, 256, 0, stream>>>(out_key_proj_w, Wok_t, DI, DM, DM);
  k_wconv<<<(DI * DM + 255) / 256, 256, 0, stream>>>(out_query_proj_w, Woq_t, DI, DM, DM);
  k_wconv<<<(DM * 8 * DM + 255) / 256, 256, 0, stream>>>(ffn_fc1_w, W1_t, DM, 8 * DM, 8 * DM);
  k_wconv<<<(4 * DM * DM + 255) / 256, 256, 0, stream>>>(ffn_fc2_w, W2_t, 4 * DM, DM, DM);

  // kp = (features+pos) @ key_proj_w   [4096 x 1040]
  gemm(fin_bf, Wkey_t, kp, (int)BL, DKEYP, DM, nullptr, nullptr);
  // init = qp @ query_proj_w           [256 x 512]
  gemm(qp_bf, Wq_t, initb, BSZ * NQ, DI, DM, nullptr, nullptr);
  // conv + silu
  { int n = BSZ * LSEQ * DCV;
    k_dwconv<<<(n + 255) / 256, 256, 0, stream>>>(kp, key_conv_w, key_conv_b, xbuf, bbias, cbias); }
  // distance MLP -> decay / dtB / C
  { int n = BSZ * LSEQ * NQ;
    k_dist<<<(n + 255) / 256, 256, 0, stream>>>(anchors, key_pos, kp, bbias, cbias,
        dist_w1, dist_b1, dist_w2, bc_proj_w, dt_proj_w, dt_bias, A_log,
        Cssm, decay, dtB); }
  // bidirectional scan: 32 workgroups (dir,b,h), TDM double-buffered coefficients
  k_scan<<<32, 256, 0, stream>>>(decay, dtB, Cssm, xbuf, initb, ysf, ysb, hstate);
  // gate + RMSNorm
  k_gate<<<(int)BL, 256, 0, stream>>>(ysf, ysb, xbuf, kp, Dvec, key_norm_w, g_bf);
  // out_feat = features + g @ out_key_proj_w
  gemm(g_bf, Wok_t, out_feat, (int)BL, DM, DI, nullptr, features);
  // hs -> LN -> bf16
  k_hs_ln<<<BSZ * NQ, 256, 0, stream>>>(hstate, query_norm_w, query_norm_b, lnhs_bf);
  // out_q_pre = queries + ln(hs) @ out_query_proj_w
  gemm(lnhs_bf, Woq_t, outq_pre, BSZ * NQ, DM, DI, nullptr, queries);
  // FFN
  k_ln256<<<BSZ * NQ, 256, 0, stream>>>(outq_pre, ffn_norm_w, ffn_norm_b, fn_bf);
  gemm(fn_bf, W1_t, t1, BSZ * NQ, 8 * DM, DM, ffn_fc1_b, nullptr);
  { int n = BSZ * NQ * 4 * DM;
    k_gelumul<<<(n + 255) / 256, 256, 0, stream>>>(t1, act_bf); }
  gemm(act_bf, W2_t, out_q, BSZ * NQ, DM, 4 * DM, ffn_fc2_b, outq_pre);
}
